// DeepSeekMoE_61890478735805
// MI455X (gfx1250) — compile-verified, matrix operand order checked by reading, not run on hardware
//
#include <hip/hip_runtime.h>
#include <stdint.h>

#define H_DIM 1024
#define I_DIM 1024
#define NEXP  8
#define TOPK  2
#define TOK   16384          // 4 * 4096 tokens
#define MOE_SCALE 1.0f

#define BM 64
#define BN 128
#define BK 32
#define LDSS 40              // LDS row stride in bf16 elems (80B: 16B-aligned, conflict-friendly)

typedef __bf16 bf16;
typedef __attribute__((ext_vector_type(16))) __bf16 bf16x16;
typedef __attribute__((ext_vector_type(8)))  float  f32x8;

union AFrag  { bf16x16 v; uint32_t u[8]; };
union BChunk { uint4 q; bf16 h[8]; };

// silu via v_rcp_f32 instead of IEEE divide (bf16 output rounding hides the ULP)
__device__ __forceinline__ float silu_f(float x) {
    return x * __builtin_amdgcn_rcpf(1.0f + __expf(-x));
}

// K index of 2-element pair p for the 16x32 bf16 A/B fragment layout (ISA 7.12.2)
__device__ __forceinline__ int frag_k(int p, int half) {
    return (p < 4) ? (half * 8 + 2 * p) : (16 + half * 8 + 2 * (p - 4));
}

__device__ __forceinline__ void load_frag(const bf16* __restrict__ base, int half, AFrag& f) {
#pragma unroll
    for (int p = 0; p < 8; ++p)
        f.u[p] = *reinterpret_cast<const uint32_t*>(base + frag_k(p, half));
}

// ---------------------------------------------------------------- conversion
__global__ void cvt_f32_bf16(const float* __restrict__ s, bf16* __restrict__ d, int n4) {
    int i = blockIdx.x * blockDim.x + threadIdx.x;
    if (i >= n4) return;
    float4 v = reinterpret_cast<const float4*>(s)[i];
    bf16* o = d + (size_t)i * 4;
    o[0] = (bf16)v.x; o[1] = (bf16)v.y; o[2] = (bf16)v.z; o[3] = (bf16)v.w;
}

// ---------------------------------------------------------------- router
__global__ void router_kernel(const float* __restrict__ x,
                              const float* __restrict__ wr,
                              float* __restrict__ logits_out,
                              int*   __restrict__ counts,
                              int*   __restrict__ top_idx,
                              float* __restrict__ top_w) {
    const int wave = (blockIdx.x * blockDim.x + threadIdx.x) >> 5;
    const int lane = threadIdx.x & 31;
    if (wave >= TOK) return;
    const float* xr = x + (size_t)wave * H_DIM;

    float acc[NEXP];
#pragma unroll
    for (int e = 0; e < NEXP; ++e) acc[e] = 0.f;
    for (int h = lane; h < H_DIM; h += 32) {
        float xv = xr[h];
        const float* wrow = wr + (size_t)h * NEXP;
#pragma unroll
        for (int e = 0; e < NEXP; ++e) acc[e] += xv * wrow[e];
    }
#pragma unroll
    for (int e = 0; e < NEXP; ++e)
        for (int m = 16; m > 0; m >>= 1) acc[e] += __shfl_xor(acc[e], m, 32);

    if (lane == 0) {
        float mx = acc[0];
#pragma unroll
        for (int e = 1; e < NEXP; ++e) mx = fmaxf(mx, acc[e]);
        float p[NEXP]; float se = 0.f;
#pragma unroll
        for (int e = 0; e < NEXP; ++e) { p[e] = __expf(acc[e] - mx); se += p[e]; }
        float inv = __builtin_amdgcn_rcpf(se);
#pragma unroll
        for (int e = 0; e < NEXP; ++e) p[e] *= inv;

        float* lrow = logits_out + (size_t)wave * NEXP;
#pragma unroll
        for (int e = 0; e < NEXP; ++e) lrow[e] = acc[e];

        int i0 = 0; float v0 = p[0];
        for (int e = 1; e < NEXP; ++e) if (p[e] > v0) { v0 = p[e]; i0 = e; }
        int i1 = -1; float v1 = -1.f;
        for (int e = 0; e < NEXP; ++e) if (e != i0 && p[e] > v1) { v1 = p[e]; i1 = e; }

        top_idx[wave * 2 + 0] = i0; top_w[wave * 2 + 0] = v0;
        top_idx[wave * 2 + 1] = i1; top_w[wave * 2 + 1] = v1;
        atomicAdd(&counts[i0], 1);
        atomicAdd(&counts[i1], 1);
    }
}

__global__ void zero_counts_kernel(int* __restrict__ counts) {
    if (threadIdx.x < NEXP) counts[threadIdx.x] = 0;
}

__global__ void offsets_kernel(const int* __restrict__ counts,
                               int* __restrict__ offsets, int* __restrict__ cursors) {
    if (threadIdx.x == 0 && blockIdx.x == 0) {
        int o = 0;
        for (int e = 0; e < NEXP; ++e) { offsets[e] = o; o += counts[e]; cursors[e] = 0; }
    }
}

__global__ void scatter_kernel(const int* __restrict__ top_idx, const float* __restrict__ top_w,
                               const int* __restrict__ offsets, int* __restrict__ cursors,
                               int* __restrict__ slot_token, float* __restrict__ slot_weight) {
    int t = blockIdx.x * blockDim.x + threadIdx.x;
    if (t >= TOK) return;
#pragma unroll
    for (int k = 0; k < TOPK; ++k) {
        int e = top_idx[t * TOPK + k];
        int slot = offsets[e] + atomicAdd(&cursors[e], 1);
        slot_token[slot]  = t;
        slot_weight[slot] = top_w[t * TOPK + k];
    }
}

// ---------------------------------------------------------------- fused gate+up GEMM
template <bool ROUTED>
__launch_bounds__(256)
__global__ void moe_gemm_gateup(const bf16* __restrict__ X,
                                const bf16* __restrict__ WgBase,
                                const bf16* __restrict__ WuBase,
                                bf16* __restrict__ Hid,
                                const int* __restrict__ slot_token,
                                const int* __restrict__ seg_off,
                                const int* __restrict__ seg_cnt) {
    __shared__ bf16 As [BM * LDSS];
    __shared__ bf16 Bgs[BN * LDSS];
    __shared__ bf16 Bus[BN * LDSS];

    const int e        = blockIdx.z;
    const int segStart = ROUTED ? seg_off[e] : 0;
    const int Me       = ROUTED ? seg_cnt[e] : TOK;
    const int mBase    = blockIdx.y * BM;
    if (ROUTED && mBase >= Me) return;
    const int nBase = blockIdx.x * BN;

    const size_t wOff = ROUTED ? (size_t)e * H_DIM * I_DIM : 0;
    const bf16* __restrict__ Wg = WgBase + wOff;
    const bf16* __restrict__ Wu = WuBase + wOff;

    const int tid = threadIdx.x;
    const int arow = tid >> 2, achunk = tid & 3;
    const int m_glob = mBase + arow;
    int src_row;
    if constexpr (ROUTED) {
        const bool a_valid = (m_glob < Me);
        src_row = slot_token[segStart + (a_valid ? m_glob : mBase)];
    } else {
        src_row = m_glob;
    }
    const bf16* __restrict__ aptr = X + (size_t)src_row * H_DIM + achunk * 8;

    const int lane = tid & 31, wv = tid >> 5;
    const int wm = wv >> 2, wn = wv & 3;
    const int half = lane >> 4, l16 = lane & 15;

    f32x8 acc_g[2][2], acc_u[2][2];
    const f32x8 zv = {0.f,0.f,0.f,0.f,0.f,0.f,0.f,0.f};
#pragma unroll
    for (int i = 0; i < 2; ++i)
#pragma unroll
        for (int j = 0; j < 2; ++j) { acc_g[i][j] = zv; acc_u[i][j] = zv; }

    for (int k0 = 0; k0 < H_DIM; k0 += BK) {
        uint4 av = *reinterpret_cast<const uint4*>(aptr + k0);
        BChunk tg[2], tu[2];
#pragma unroll
        for (int i = 0; i < 2; ++i) {
            const int c = tid + i * 256, krow = c >> 4, nch = c & 15;
            tg[i].q = *reinterpret_cast<const uint4*>(Wg + (size_t)(k0 + krow) * I_DIM + nBase + nch * 8);
            tu[i].q = *reinterpret_cast<const uint4*>(Wu + (size_t)(k0 + krow) * I_DIM + nBase + nch * 8);
        }
        __syncthreads();
        *reinterpret_cast<uint4*>(&As[arow * LDSS + achunk * 8]) = av;
#pragma unroll
        for (int i = 0; i < 2; ++i) {
            const int c = tid + i * 256, krow = c >> 4, nch = c & 15;
#pragma unroll
            for (int j = 0; j < 8; ++j) {
                Bgs[(nch * 8 + j) * LDSS + krow] = tg[i].h[j];   // transposed: [n][k]
                Bus[(nch * 8 + j) * LDSS + krow] = tu[i].h[j];
            }
        }
        __syncthreads();

        AFrag af[2], bg[2], bu[2];
#pragma unroll
        for (int tm = 0; tm < 2; ++tm)
            load_frag(&As[(wm * 32 + tm * 16 + l16) * LDSS], half, af[tm]);
#pragma unroll
        for (int tn = 0; tn < 2; ++tn) {
            load_frag(&Bgs[(wn * 32 + tn * 16 + l16) * LDSS], half, bg[tn]);
            load_frag(&Bus[(wn * 32 + tn * 16 + l16) * LDSS], half, bu[tn]);
        }
#pragma unroll
        for (int tm = 0; tm < 2; ++tm)
#pragma unroll
            for (int tn = 0; tn < 2; ++tn) {
                acc_g[tm][tn] = __builtin_amdgcn_wmma_f32_16x16x32_bf16(
                    false, af[tm].v, false, bg[tn].v, (short)0, acc_g[tm][tn], false, false);
                acc_u[tm][tn] = __builtin_amdgcn_wmma_f32_16x16x32_bf16(
                    false, af[tm].v, false, bu[tn].v, (short)0, acc_u[tm][tn], false, false);
            }
    }

#pragma unroll
    for (int tm = 0; tm < 2; ++tm)
#pragma unroll
        for (int tn = 0; tn < 2; ++tn)
#pragma unroll
            for (int r = 0; r < 8; ++r) {
                const int m = mBase + wm * 32 + tm * 16 + half * 8 + r;
                if (!ROUTED || m < Me) {
                    const int n = nBase + wn * 32 + tn * 16 + l16;
                    float hv = silu_f(acc_g[tm][tn][r]) * acc_u[tm][tn][r];
                    Hid[(size_t)(segStart + m) * I_DIM + n] = (bf16)hv;
                }
            }
}

// ---------------------------------------------------------------- down-proj GEMM
template <bool ROUTED>
__launch_bounds__(256)
__global__ void moe_gemm_down(const bf16* __restrict__ HidBase,
                              const bf16* __restrict__ WdBase,
                              float* __restrict__ Out,
                              const int*   __restrict__ slot_token,
                              const float* __restrict__ slot_weight,
                              const int*   __restrict__ seg_off,
                              const int*   __restrict__ seg_cnt) {
    __shared__ bf16 As[BM * LDSS];
    __shared__ bf16 Bs[BN * LDSS];

    const int e        = blockIdx.z;
    const int segStart = ROUTED ? seg_off[e] : 0;
    const int Me       = ROUTED ? seg_cnt[e] : TOK;
    const int mBase    = blockIdx.y * BM;
    if (ROUTED && mBase >= Me) return;
    const int nBase = blockIdx.x * BN;

    const bf16* __restrict__ Wd = WdBase + (ROUTED ? (size_t)e * I_DIM * H_DIM : 0);

    const int tid = threadIdx.x;
    const int arow = tid >> 2, achunk = tid & 3;
    const int m_glob = mBase + arow;
    int a_row;
    if constexpr (ROUTED) {
        const bool a_valid = (m_glob < Me);
        a_row = segStart + (a_valid ? m_glob : mBase);
    } else {
        a_row = m_glob;
    }
    const bf16* __restrict__ aptr = HidBase + (size_t)a_row * I_DIM + achunk * 8;

    const int lane = tid & 31, wv = tid >> 5;
    const int wm = wv >> 2, wn = wv & 3;
    const int half = lane >> 4, l16 = lane & 15;

    f32x8 acc[2][2];
    const f32x8 zv = {0.f,0.f,0.f,0.f,0.f,0.f,0.f,0.f};
#pragma unroll
    for (int i = 0; i < 2; ++i)
#pragma unroll
        for (int j = 0; j < 2; ++j) acc[i][j] = zv;

    for (int k0 = 0; k0 < I_DIM; k0 += BK) {
        uint4 av = *reinterpret_cast<const uint4*>(aptr + k0);
        BChunk tb[2];
#pragma unroll
        for (int i = 0; i < 2; ++i) {
            const int c = tid + i * 256, krow = c >> 4, nch = c & 15;
            tb[i].q = *reinterpret_cast<const uint4*>(Wd + (size_t)(k0 + krow) * H_DIM + nBase + nch * 8);
        }
        __syncthreads();
        *reinterpret_cast<uint4*>(&As[arow * LDSS + achunk * 8]) = av;
#pragma unroll
        for (int i = 0; i < 2; ++i) {
            const int c = tid + i * 256, krow = c >> 4, nch = c & 15;
#pragma unroll
            for (int j = 0; j < 8; ++j)
                Bs[(nch * 8 + j) * LDSS + krow] = tb[i].h[j];
        }
        __syncthreads();

        AFrag af[2], bf[2];
#pragma unroll
        for (int tm = 0; tm < 2; ++tm)
            load_frag(&As[(wm * 32 + tm * 16 + l16) * LDSS], half, af[tm]);
#pragma unroll
        for (int tn = 0; tn < 2; ++tn)
            load_frag(&Bs[(wn * 32 + tn * 16 + l16) * LDSS], half, bf[tn]);
#pragma unroll
        for (int tm = 0; tm < 2; ++tm)
#pragma unroll
            for (int tn = 0; tn < 2; ++tn)
                acc[tm][tn] = __builtin_amdgcn_wmma_f32_16x16x32_bf16(
                    false, af[tm].v, false, bf[tn].v, (short)0, acc[tm][tn], false, false);
    }

#pragma unroll
    for (int tm = 0; tm < 2; ++tm)
#pragma unroll
        for (int tn = 0; tn < 2; ++tn)
#pragma unroll
            for (int r = 0; r < 8; ++r) {
                const int m = mBase + wm * 32 + tm * 16 + half * 8 + r;
                const int n = nBase + wn * 32 + tn * 16 + l16;
                const float v = acc[tm][tn][r];
                if constexpr (ROUTED) {
                    if (m < Me) {
                        const int   t = slot_token [segStart + m];
                        const float w = slot_weight[segStart + m];
                        atomicAdd(&Out[(size_t)t * H_DIM + n], MOE_SCALE * w * v);
                    }
                } else {
                    Out[(size_t)m * H_DIM + n] = v;
                }
            }
}

// ---------------------------------------------------------------- launch
extern "C" void kernel_launch(void* const* d_in, const int* in_sizes, int n_in,
                              void* d_out, int out_size, void* d_ws, size_t ws_size,
                              hipStream_t stream) {
    const float* x    = (const float*)d_in[0];
    const float* wr   = (const float*)d_in[1];
    const float* wshg = (const float*)d_in[2];
    const float* wshu = (const float*)d_in[3];
    const float* wshd = (const float*)d_in[4];
    const float* wg   = (const float*)d_in[5];
    const float* wu   = (const float*)d_in[6];
    const float* wd   = (const float*)d_in[7];

    float* out    = (float*)d_out;
    float* logits = out + (size_t)TOK * H_DIM;

    char* p = (char*)d_ws;
    auto alloc = [&](size_t bytes) -> char* {
        char* r = p; p += (bytes + 255) & ~(size_t)255; return r;
    };
    bf16*  xb      = (bf16*) alloc((size_t)TOK * H_DIM * 2);
    bf16*  wshg_b  = (bf16*) alloc((size_t)H_DIM * I_DIM * 2);
    bf16*  wshu_b  = (bf16*) alloc((size_t)H_DIM * I_DIM * 2);
    bf16*  wshd_b  = (bf16*) alloc((size_t)I_DIM * H_DIM * 2);
    bf16*  wg_b    = (bf16*) alloc((size_t)NEXP * H_DIM * I_DIM * 2);
    bf16*  wu_b    = (bf16*) alloc((size_t)NEXP * H_DIM * I_DIM * 2);
    bf16*  wd_b    = (bf16*) alloc((size_t)NEXP * I_DIM * H_DIM * 2);
    bf16*  hid_sh  = (bf16*) alloc((size_t)TOK * I_DIM * 2);
    bf16*  hid_rt  = (bf16*) alloc((size_t)2 * TOK * I_DIM * 2);
    int*   top_idx = (int*)  alloc((size_t)TOK * TOPK * 4);
    float* top_w   = (float*)alloc((size_t)TOK * TOPK * 4);
    int*   slot_tk = (int*)  alloc((size_t)2 * TOK * 4);
    float* slot_w  = (float*)alloc((size_t)2 * TOK * 4);
    int*   counts  = (int*)  alloc(256);
    int*   offsets = (int*)  alloc(256);
    int*   cursors = (int*)  alloc(256);

    auto cvt = [&](const float* s, bf16* dst, size_t n) {
        int n4 = (int)(n / 4);
        cvt_f32_bf16<<<(n4 + 255) / 256, 256, 0, stream>>>(s, dst, n4);
    };
    cvt(x,    xb,     (size_t)TOK * H_DIM);
    cvt(wshg, wshg_b, (size_t)H_DIM * I_DIM);
    cvt(wshu, wshu_b, (size_t)H_DIM * I_DIM);
    cvt(wshd, wshd_b, (size_t)I_DIM * H_DIM);
    cvt(wg,   wg_b,   (size_t)NEXP * H_DIM * I_DIM);
    cvt(wu,   wu_b,   (size_t)NEXP * H_DIM * I_DIM);
    cvt(wd,   wd_b,   (size_t)NEXP * I_DIM * H_DIM);

    zero_counts_kernel<<<1, 32, 0, stream>>>(counts);
    router_kernel<<<TOK / 8, 256, 0, stream>>>(x, wr, logits, counts, top_idx, top_w);
    offsets_kernel<<<1, 1, 0, stream>>>(counts, offsets, cursors);
    scatter_kernel<<<TOK / 256, 256, 0, stream>>>(top_idx, top_w, offsets, cursors, slot_tk, slot_w);

    dim3 blk(256);
    // shared expert: gate+up fused, then down writes d_out (initializes it)
    moe_gemm_gateup<false><<<dim3(I_DIM / BN, TOK / BM, 1), blk, 0, stream>>>(
        xb, wshg_b, wshu_b, hid_sh, slot_tk, offsets, counts);
    moe_gemm_down<false><<<dim3(H_DIM / BN, TOK / BM, 1), blk, 0, stream>>>(
        hid_sh, wshd_b, out, slot_tk, slot_w, offsets, counts);
    // routed experts: gathered grouped GEMMs over top-2 assignments (2T rows total)
    moe_gemm_gateup<true><<<dim3(I_DIM / BN, TOK / BM, NEXP), blk, 0, stream>>>(
        xb, wg_b, wu_b, hid_rt, slot_tk, offsets, counts);
    moe_gemm_down<true><<<dim3(H_DIM / BN, TOK / BM, NEXP), blk, 0, stream>>>(
        hid_rt, wd_b, out, slot_tk, slot_w, offsets, counts);
}
